// GCN_3l_5025111736761
// MI455X (gfx1250) — matrix operand order repaired and checked by reading, not compiled, not run on hardware
//
#include <hip/hip_runtime.h>
#include <hip/hip_bf16.h>

typedef _Float16 half_t;
typedef __attribute__((ext_vector_type(16))) _Float16 v16h;
typedef __attribute__((ext_vector_type(8)))  _Float16 v8h;
typedef __attribute__((ext_vector_type(8)))  float    v8f;

#define HDIM 128

// ---------------- small utility kernels ----------------

__global__ void k_fill(float* __restrict__ p, float v, int n) {
    int i = blockIdx.x * blockDim.x + threadIdx.x;
    if (i < n) p[i] = v;
}

// deg[col[e]] += 1  (deg pre-initialized to 1.0 for the self loop)
__global__ void k_deg(const int* __restrict__ ei, float* __restrict__ deg, int E) {
    int e = blockIdx.x * blockDim.x + threadIdx.x;
    if (e < E) atomicAdd(&deg[ei[E + e]], 1.0f);
}

__global__ void k_rsqrt(float* __restrict__ p, int n) {
    int i = blockIdx.x * blockDim.x + threadIdx.x;
    if (i < n) p[i] = rsqrtf(p[i]);   // deg >= 1 always (self loop)
}

// W [128,128] f32 row-major -> Wt f16 with Wt[n*128+k] = W[k*128+n]
__global__ void k_wcvt(const float* __restrict__ W, half_t* __restrict__ Wt) {
    int t = blockIdx.x * blockDim.x + threadIdx.x;   // t = k*128 + n
    if (t < HDIM * HDIM) {
        int k = t >> 7, n = t & 127;
        Wt[n * HDIM + k] = (half_t)W[t];
    }
}

// ---------------- WMMA GEMM: C[nrows,128] = A[nrows,128] @ W ----------------
// A fp32 row-major, weight pre-transposed to f16 (Wt[n][k]), C fp32.
// One block = 128 rows (8 waves x 16 rows), each wave owns a 16x128 strip.
__global__ void __launch_bounds__(256)
k_gemm(const float* __restrict__ A, const half_t* __restrict__ Bt,
       float* __restrict__ C, int nrows) {
    const int wave = threadIdx.x >> 5;
    const int lane = threadIdx.x & 31;
    const int m    = lane & 15;
    const bool hi  = lane >= 16;
    const int rowBase = blockIdx.x * 128 + wave * 16;

    int r = rowBase + m;
    const int rr = (r < nrows) ? r : (nrows - 1);        // clamp loads, guard stores
    const float* ap0 = A + (size_t)rr * HDIM + (hi ? 8 : 0);

    v8f acc[8];
#pragma unroll
    for (int c = 0; c < 8; ++c)
#pragma unroll
        for (int j = 0; j < 8; ++j) acc[c][j] = 0.0f;

#pragma unroll
    for (int k0 = 0; k0 < HDIM; k0 += 32) {
        // A 16x32 f16 fragment (ISA layout): lane<16 -> K {k0+0..7, k0+16..23},
        // lane>=16 -> K {k0+8..15, k0+24..31}; row M = lane%16.
        const float4 a0 = *(const float4*)(ap0 + k0);
        const float4 a1 = *(const float4*)(ap0 + k0 + 4);
        const float4 a2 = *(const float4*)(ap0 + k0 + 16);
        const float4 a3 = *(const float4*)(ap0 + k0 + 20);
        v16h a;
        a[0]  = (half_t)a0.x; a[1]  = (half_t)a0.y; a[2]  = (half_t)a0.z; a[3]  = (half_t)a0.w;
        a[4]  = (half_t)a1.x; a[5]  = (half_t)a1.y; a[6]  = (half_t)a1.z; a[7]  = (half_t)a1.w;
        a[8]  = (half_t)a2.x; a[9]  = (half_t)a2.y; a[10] = (half_t)a2.z; a[11] = (half_t)a2.w;
        a[12] = (half_t)a3.x; a[13] = (half_t)a3.y; a[14] = (half_t)a3.z; a[15] = (half_t)a3.w;

#pragma unroll
        for (int c = 0; c < 8; ++c) {
            // B 32x16 fragment: col N = lane%16, K = k0 + (hi?8:0) + {0..7, 16..23},
            // matching the A fragment layout in the transposed weight.
            const half_t* bp = Bt + (size_t)(c * 16 + m) * HDIM + k0 + (hi ? 8 : 0);
            const v8h b0 = *(const v8h*)(bp);
            const v8h b1 = *(const v8h*)(bp + 16);
            v16h b;
#pragma unroll
            for (int i = 0; i < 8; ++i) { b[i] = b0[i]; b[i + 8] = b1[i]; }

            acc[c] = __builtin_amdgcn_wmma_f32_16x16x32_f16(
                false, a, false, b, (short)0, acc[c], false, false);
        }
    }

    // C/D layout: VGPR j -> row (rowBase + j + (hi?8:0)), col = c*16 + lane%16
#pragma unroll
    for (int c = 0; c < 8; ++c) {
        const int col = c * 16 + m;
#pragma unroll
        for (int j = 0; j < 8; ++j) {
            const int row = rowBase + j + (hi ? 8 : 0);
            if (row < nrows) C[(size_t)row * HDIM + col] = acc[c][j];
        }
    }
}

// ---------------- aggregation ----------------

// out = t * dinv[node]^2   (self-loop contribution doubles as zero-init)
__global__ void k_agg_init(const float* __restrict__ t, const float* __restrict__ dinv,
                           float* __restrict__ out, int total) {
    int i = blockIdx.x * blockDim.x + threadIdx.x;
    if (i < total) {
        float d = dinv[i >> 7];
        out[i] = t[i] * d * d;
    }
}

// one wave per edge, 4 features per lane: out[col] += t[row] * dinv[row]*dinv[col]
__global__ void k_agg_edges(const float* __restrict__ t, const int* __restrict__ ei,
                            const float* __restrict__ dinv, float* __restrict__ out, int E) {
    int gid  = blockIdx.x * blockDim.x + threadIdx.x;
    int e    = gid >> 5;
    int lane = gid & 31;
    if (e < E) {
        int r = ei[e];
        int c = ei[E + e];
        float w = dinv[r] * dinv[c];
        const float4 v = *(const float4*)(t + (size_t)r * HDIM + lane * 4);
        float* o = out + (size_t)c * HDIM + lane * 4;
        atomicAdd(o + 0, v.x * w);
        atomicAdd(o + 1, v.y * w);
        atomicAdd(o + 2, v.z * w);
        atomicAdd(o + 3, v.w * w);
    }
}

__global__ void k_bias_relu(float* __restrict__ h, const float* __restrict__ b) {
    size_t i = (size_t)blockIdx.x * HDIM + threadIdx.x;
    float v = h[i] + b[threadIdx.x];
    h[i] = v > 0.0f ? v : 0.0f;
}

// ---------------- pooling + FFN head ----------------

__global__ void k_pool(const float* __restrict__ h, const int* __restrict__ batch,
                       float* __restrict__ pooled, float* __restrict__ cnt) {
    int node = blockIdx.x;
    int f = threadIdx.x;
    int g = batch[node];
    atomicAdd(&pooled[(size_t)g * HDIM + f], h[(size_t)node * HDIM + f]);
    if (f == 0) atomicAdd(&cnt[g], 1.0f);
}

__global__ void k_pool_div(float* __restrict__ pooled, const float* __restrict__ cnt) {
    int g = blockIdx.x, f = threadIdx.x;
    pooled[(size_t)g * HDIM + f] /= fmaxf(cnt[g], 1.0f);
}

// out[g,c] = bf2[c] + sum_h z[g,h] * Wf2[h,c]   (G=128, C=2 -> 256 threads)
__global__ void k_ffn_out(const float* __restrict__ z, const float* __restrict__ Wf2,
                          const float* __restrict__ bf2, float* __restrict__ out) {
    int t = threadIdx.x;
    int g = t >> 1, c = t & 1;
    float acc = bf2[c];
#pragma unroll 8
    for (int h = 0; h < HDIM; ++h) acc += z[(size_t)g * HDIM + h] * Wf2[h * 2 + c];
    out[g * 2 + c] = acc;
}

// ---------------- host launch ----------------

extern "C" void kernel_launch(void* const* d_in, const int* in_sizes, int n_in,
                              void* d_out, int out_size, void* d_ws, size_t ws_size,
                              hipStream_t stream) {
    const float* x     = (const float*)d_in[0];
    const int*   ei    = (const int*)d_in[1];
    const int*   batch = (const int*)d_in[2];
    const float* W1 = (const float*)d_in[3];  const float* b1  = (const float*)d_in[4];
    const float* W2 = (const float*)d_in[5];  const float* b2  = (const float*)d_in[6];
    const float* W3 = (const float*)d_in[7];  const float* b3  = (const float*)d_in[8];
    const float* Wf1 = (const float*)d_in[9]; const float* bf1 = (const float*)d_in[10];
    const float* Wf2 = (const float*)d_in[11];const float* bf2 = (const float*)d_in[12];
    float* out = (float*)d_out;

    const int N = in_sizes[0] / HDIM;
    const int E = in_sizes[1] / 2;
    const int G = 128;

    // workspace carve (256B aligned)
    char* w = (char*)d_ws;
    auto carve = [&](size_t bytes) { char* p = w; w += (bytes + 255) & ~(size_t)255; return p; };
    float*  dinv   = (float*)carve((size_t)N * 4);
    float*  bufA   = (float*)carve((size_t)N * HDIM * 4);
    float*  bufB   = (float*)carve((size_t)N * HDIM * 4);
    half_t* Wt     = (half_t*)carve((size_t)HDIM * HDIM * 2);
    float*  pooled = (float*)carve((size_t)G * HDIM * 4);
    float*  cnt    = (float*)carve((size_t)G * 4);

    // normalization: deg = 1 + in-degree; dinv = rsqrt(deg)
    k_fill<<<(N + 255) / 256, 256, 0, stream>>>(dinv, 1.0f, N);
    k_deg<<<(E + 255) / 256, 256, 0, stream>>>(ei, dinv, E);
    k_rsqrt<<<(N + 255) / 256, 256, 0, stream>>>(dinv, N);

    const float* Ws[3] = {W1, W2, W3};
    const float* bs[3] = {b1, b2, b3};
    const float* hin = x;
    const int gemmGrid = (N + 127) / 128;
    const int totalNH  = N * HDIM;

    for (int l = 0; l < 3; ++l) {
        k_wcvt<<<64, 256, 0, stream>>>(Ws[l], Wt);
        k_gemm<<<gemmGrid, 256, 0, stream>>>(hin, Wt, bufA, N);
        k_agg_init<<<(totalNH + 255) / 256, 256, 0, stream>>>(bufA, dinv, bufB, totalNH);
        k_agg_edges<<<(E * 32 + 255) / 256, 256, 0, stream>>>(bufA, ei, dinv, bufB, E);
        k_bias_relu<<<N, HDIM, 0, stream>>>(bufB, bs[l]);
        hin = bufB;
    }

    // global mean pool
    k_fill<<<(G * HDIM + 255) / 256, 256, 0, stream>>>(pooled, 0.0f, G * HDIM);
    k_fill<<<1, 256, 0, stream>>>(cnt, 0.0f, G);
    k_pool<<<N, HDIM, 0, stream>>>(bufB, batch, pooled, cnt);
    k_pool_div<<<G, HDIM, 0, stream>>>(pooled, cnt);

    // FFN head: relu(pooled@Wf1+bf1) @ Wf2 + bf2
    k_wcvt<<<64, 256, 0, stream>>>(Wf1, Wt);
    k_gemm<<<1, 256, 0, stream>>>(pooled, Wt, bufA, G);   // bufA[0:128] = z
    k_bias_relu<<<G, HDIM, 0, stream>>>(bufA, bf1);
    k_ffn_out<<<1, 256, 0, stream>>>(bufA, Wf2, bf2, out);
}
